// VisionMambaUNet_43276090474779
// MI455X (gfx1250) — compile-verified, matrix operand order checked by reading, not run on hardware
//
// VisionMambaUNet for MI455X (gfx1250, CDNA5) — wave32 + WMMA implementation.
//
// Roofline: whole net <100MB working set (fits MI455X 192MB L2), ~20 GFLOP =>
// latency/fusion bound. Strategy: f16-in/f32-acc WMMA for every projection,
// fully fused epilogues, scan state resident in VGPRs.
// This revision eliminates all exec-predicated loads from the GEMM hot loop:
// M/N edges are handled by clamped (always-valid) addresses whose garbage only
// lands in never-stored output rows/cols; K raggedness is confined to one tail
// iteration using value-selects (v_cndmask), not masked loads. The contiguous
// A path (amode 0) is a template specialization using aligned float4 loads
// (global_load_b128), so the K loop issues wide loads back-to-back and the two
// v_wmma per step overlap with memory.

#include <hip/hip_runtime.h>
#include <hip/hip_bf16.h>
#include <math.h>

typedef __attribute__((ext_vector_type(16))) _Float16 v16h;
typedef __attribute__((ext_vector_type(8)))  float    v8f;

#define DEV __device__ __forceinline__

// ---------------------------------------------------------------- activations
DEV float act_apply(float v, int act) {
  switch (act) {
    case 1: return v > 0.f ? v : 0.f;                       // relu
    case 2: return v / (1.f + __expf(-v));                  // silu
    case 3: return v > 20.f ? v : log1pf(__expf(v));        // softplus
  }
  return v;
}

DEV void pack4(v16h& f, int base, float4 v) {
  f[base + 0] = (_Float16)v.x; f[base + 1] = (_Float16)v.y;
  f[base + 2] = (_Float16)v.z; f[base + 3] = (_Float16)v.w;
}

// ------------------------------------------------------------------ WMMA GEMM
// out[m,n] = act( sum_k A[m,k]*W[n,k] + bias[n] + resid )   (C = A * W^T).
// AMODE 0: A row-major (lda), contiguous K  -> float4 loads.
// AMODE 1: A is NCHW viewed as (B*L, K): A[m,k] = T[b,k,l], K-stride = L
//          (L multiple of 16 => batch split is tile-uniform).
// omode 0: out[m*ldo+n].  omode 1: NCHW store out[(b*N+n)*L+l] (+resid, act).
// One wave32 computes a 16x32 tile (one A fragment, two B fragments, two
// v_wmma per 32-wide K step). Edge rows/cols use clamped addresses (garbage
// only reaches never-stored outputs); only the K tail needs value zeroing.
template <int AMODE>
__global__ void k_gemm_wmma(const float* __restrict__ A, int lda, int L,
                            const float* __restrict__ W, int ldw,
                            const float* __restrict__ bias,
                            const float* __restrict__ resid,   // NCHW (omode 1)
                            float* __restrict__ out, int ldo, int omode,
                            int M, int N, int Ktot, int act) {
  const int lane = threadIdx.x & 31;
  const int wave = threadIdx.x >> 5;
  const int tiles_n = (N + 31) >> 5;              // 32-wide N tiles
  const int tiles_m = (M + 15) >> 4;
  const int tile = blockIdx.x * (blockDim.x >> 5) + wave;
  if (tile >= tiles_m * tiles_n) return;          // wave-uniform: EXEC all-1s
  const int tm = (int)((unsigned)tile / (unsigned)tiles_n);
  const int tn = tile - tm * tiles_n;

  const int row  = lane & 15;          // A row / B column within tile
  const int kgrp = (lane >> 4) << 3;   // lanes 0-15: K 0..7,16..23 ; 16-31: 8..15,24..31
  const int m0 = tm << 4;
  const int n0 = tn << 5;

  // wave-uniform batch/position split (u32 divide, once per wave)
  const int bdiv = (int)((unsigned)m0 / (unsigned)L);
  const int lrem = m0 - bdiv * L;

  // clamped (always-valid) row/col bases: no predication in the K loop
  const int mC = min(m0 + row, M - 1);
  const float* arow = (AMODE == 0)
      ? (A + (long)mC * lda)
      : (A + (long)bdiv * Ktot * L + (lrem + row));   // amode1: rows in-bounds
  const int nC0 = min(n0 + row, N - 1);
  const int nC1 = min(n0 + 16 + row, N - 1);
  const float* brow0 = W + (long)nC0 * ldw;
  const float* brow1 = W + (long)nC1 * ldw;

  v8f acc0 = {}, acc1 = {};
  const int Kmain = Ktot & ~31;
  int k0 = 0;
  for (; k0 < Kmain; k0 += 32) {                  // unconditional wide loads
    v16h a, b0, b1;
    const int kb = k0 + kgrp;
    if (AMODE == 0) {
      pack4(a, 0,  *(const float4*)(arow + kb));
      pack4(a, 4,  *(const float4*)(arow + kb + 4));
      pack4(a, 8,  *(const float4*)(arow + kb + 16));
      pack4(a, 12, *(const float4*)(arow + kb + 20));
    } else {
#pragma unroll
      for (int e = 0; e < 8; ++e) {
        a[e]     = (_Float16)arow[(long)(kb + e) * L];
        a[e + 8] = (_Float16)arow[(long)(kb + 16 + e) * L];
      }
    }
    pack4(b0, 0,  *(const float4*)(brow0 + kb));
    pack4(b0, 4,  *(const float4*)(brow0 + kb + 4));
    pack4(b0, 8,  *(const float4*)(brow0 + kb + 16));
    pack4(b0, 12, *(const float4*)(brow0 + kb + 20));
    pack4(b1, 0,  *(const float4*)(brow1 + kb));
    pack4(b1, 4,  *(const float4*)(brow1 + kb + 4));
    pack4(b1, 8,  *(const float4*)(brow1 + kb + 16));
    pack4(b1, 12, *(const float4*)(brow1 + kb + 20));
    acc0 = __builtin_amdgcn_wmma_f32_16x16x32_f16(false, a, false, b0,
                                                  (short)0, acc0, false, false);
    acc1 = __builtin_amdgcn_wmma_f32_16x16x32_f16(false, a, false, b1,
                                                  (short)0, acc1, false, false);
  }
  if (k0 < Ktot) {               // ragged K tail: clamped loads + value select
    v16h a, b0, b1;
#pragma unroll
    for (int e = 0; e < 8; ++e) {
      const int ka = k0 + kgrp + e, kb = ka + 16;
      const int kca = min(ka, Ktot - 1), kcb = min(kb, Ktot - 1);
      const bool va = ka < Ktot, vb = kb < Ktot;
      const float aa = (AMODE == 0) ? arow[kca] : arow[(long)kca * L];
      const float ab = (AMODE == 0) ? arow[kcb] : arow[(long)kcb * L];
      a[e]      = (_Float16)(va ? aa : 0.f);
      a[e + 8]  = (_Float16)(vb ? ab : 0.f);
      const float p0 = brow0[kca], q0 = brow0[kcb];
      const float p1 = brow1[kca], q1 = brow1[kcb];
      b0[e]     = (_Float16)(va ? p0 : 0.f);
      b0[e + 8] = (_Float16)(vb ? q0 : 0.f);
      b1[e]     = (_Float16)(va ? p1 : 0.f);
      b1[e + 8] = (_Float16)(vb ? q1 : 0.f);
    }
    acc0 = __builtin_amdgcn_wmma_f32_16x16x32_f16(false, a, false, b0,
                                                  (short)0, acc0, false, false);
    acc1 = __builtin_amdgcn_wmma_f32_16x16x32_f16(false, a, false, b1,
                                                  (short)0, acc1, false, false);
  }

  // fused epilogue: bias -> residual -> activation; row-major or NCHW store
#pragma unroll
  for (int nt = 0; nt < 2; ++nt) {
    const int n = n0 + (nt << 4) + (lane & 15);
    if (n >= N) continue;
#pragma unroll
    for (int i = 0; i < 8; ++i) {
      const int ml = (lane < 16) ? i : i + 8;
      const int m = m0 + ml;
      if (m >= M) continue;
      float v = nt ? acc1[i] : acc0[i];
      if (bias) v += bias[n];
      const long oidx = (omode == 0)
          ? (long)m * ldo + n
          : ((long)bdiv * N + n) * L + (lrem + ml);
      if (resid) v += resid[oidx];
      out[oidx] = act_apply(v, act);
    }
  }
}

// --------------------------------------------------- direct conv2d (+BN, ReLU)
// Input may be a channel-concat of two tensors (decoder skip path): channels
// [0,C1) from in1, [C1,C1+C2) from in2. w is OIHW with C = C1+C2.
__global__ void k_conv2d(const float* __restrict__ in1, int C1,
                         const float* __restrict__ in2, int C2,
                         const float* __restrict__ w,
                         const float* __restrict__ scale,
                         const float* __restrict__ bias,
                         float* __restrict__ out,
                         int Bn, int Cout, int Hin, int Win,
                         int Hout, int Wout, int K, int stride, int pad, int relu) {
  const unsigned total = (unsigned)Bn * Cout * Hout * Wout;
  const unsigned idx = blockIdx.x * blockDim.x + threadIdx.x;
  if (idx >= total) return;
  const int wo = (int)(idx % (unsigned)Wout); unsigned t = idx / (unsigned)Wout;
  const int ho = (int)(t % (unsigned)Hout);   t /= (unsigned)Hout;
  const int co = (int)(t % (unsigned)Cout);
  const int b  = (int)(t / (unsigned)Cout);
  const int C = C1 + C2;
  float acc = 0.f;
  for (int c = 0; c < C; ++c) {
    const float* src = (c < C1)
        ? (in1 + ((long)b * C1 + c) * Hin * Win)
        : (in2 + ((long)b * C2 + (c - C1)) * Hin * Win);
    const float* wr = w + ((long)co * C + c) * K * K;
    for (int kh = 0; kh < K; ++kh) {
      const int hi = ho * stride - pad + kh;
      if (hi < 0 || hi >= Hin) continue;
      for (int kw = 0; kw < K; ++kw) {
        const int wi = wo * stride - pad + kw;
        if (wi < 0 || wi >= Win) continue;
        acc += src[hi * Win + wi] * wr[kh * K + kw];
      }
    }
  }
  float v = acc * (scale ? scale[co] : 1.f) + (bias ? bias[co] : 0.f);
  if (relu) v = v > 0.f ? v : 0.f;
  out[idx] = v;
}

// --------------------------------------------- transposed conv 2x2 stride 2
// out[b,o,2h+k,2w+l] = sum_c x[b,c,h,w] * up_w[c,o,k,l]
__global__ void k_upconv2x2(const float* __restrict__ x, const float* __restrict__ w,
                            float* __restrict__ out, int Bn, int Ci, int Co,
                            int H, int W) {
  const int Ho = 2 * H, Wo = 2 * W;
  const unsigned total = (unsigned)Bn * Co * Ho * Wo;
  const unsigned idx = blockIdx.x * blockDim.x + threadIdx.x;
  if (idx >= total) return;
  const int wo = (int)(idx % (unsigned)Wo); unsigned t = idx / (unsigned)Wo;
  const int ho = (int)(t % (unsigned)Ho);   t /= (unsigned)Ho;
  const int co = (int)(t % (unsigned)Co);
  const int b  = (int)(t / (unsigned)Co);
  const int h = ho >> 1, k = ho & 1, ww = wo >> 1, l = wo & 1;
  float acc = 0.f;
  for (int c = 0; c < Ci; ++c)
    acc += x[((long)b * Ci + c) * H * W + h * W + ww] *
           w[(((long)c * Co + co) * 2 + k) * 2 + l];
  out[idx] = acc;
}

// ---------------------------------- Mamba causal depthwise conv1d + bias + SiLU
// Reads first di channels of xz (row stride 2*di); xh[b,l,e] contiguous out.
__global__ void k_dwconv_silu(const float* __restrict__ xz,
                              const float* __restrict__ w,
                              const float* __restrict__ bias,
                              float* __restrict__ xh, int Bn, int L, int di) {
  const unsigned total = (unsigned)Bn * L * di;
  const unsigned idx = blockIdx.x * blockDim.x + threadIdx.x;
  if (idx >= total) return;
  const int e = (int)(idx % (unsigned)di); unsigned t = idx / (unsigned)di;
  const int l = (int)(t % (unsigned)L);
  const int b = (int)(t / (unsigned)L);
  const long rs = 2 * (long)di;
  float acc = bias[e];
#pragma unroll
  for (int j = 0; j < 4; ++j) {
    const int ls = l - 3 + j;
    if (ls >= 0) acc += xz[((long)b * L + ls) * rs + e] * w[e * 4 + j];
  }
  xh[idx] = acc / (1.f + __expf(-acc));  // silu
}

// ----------------------------------------------------- selective scan (Mamba)
// One wave32 per (b, channel e). Lane n owns states n and n+32 (N=64).
// Per step: h = exp(dt*A)*h + (dt*x)*B ; y = sum_n h*C (shfl_xor butterfly).
// Fused epilogue: y = (y + x*D) * silu(z).
__global__ void k_scan(const float* __restrict__ xh, const float* __restrict__ dt,
                       const float* __restrict__ dbc, const float* __restrict__ xz,
                       const float* __restrict__ A_log, const float* __restrict__ Dp,
                       float* __restrict__ y, int Bn, int L, int di, int dr) {
  const int wave = (int)((blockIdx.x * blockDim.x + threadIdx.x) >> 5);
  const int lane = threadIdx.x & 31;
  if (wave >= Bn * di) return;
  const int b = (int)((unsigned)wave / (unsigned)di);
  const int e = wave - b * di;
  const int ldd = dr + 128;
  const float A0 = -__expf(A_log[e * 64 + lane]);
  const float A1 = -__expf(A_log[e * 64 + 32 + lane]);
  const float Dv = Dp[e];
  float h0 = 0.f, h1 = 0.f;
  const long base = (long)b * L;
  for (int l = 0; l < L; ++l) {
    const long r = base + l;
    const float xv  = xh[r * di + e];
    const float dtv = dt[r * di + e];
    const float* bc = dbc + r * ldd + dr;
    const float Bt0 = bc[lane],      Bt1 = bc[32 + lane];
    const float Ct0 = bc[64 + lane], Ct1 = bc[96 + lane];
    const float s = dtv * xv;
    h0 = __expf(dtv * A0) * h0 + s * Bt0;
    h1 = __expf(dtv * A1) * h1 + s * Bt1;
    float yv = h0 * Ct0 + h1 * Ct1;
#pragma unroll
    for (int off = 16; off > 0; off >>= 1) yv += __shfl_xor(yv, off, 32);
    if (lane == 0) {
      const float zv = xz[r * 2 * di + di + e];
      y[r * di + e] = (yv + xv * Dv) * (zv / (1.f + __expf(-zv)));
    }
  }
}

// ----------------------------- bilinear 2x upsample fused with 1x1 seg conv
__global__ void k_resize_seg(const float* __restrict__ x, const float* __restrict__ w,
                             const float* __restrict__ bias, float* __restrict__ out,
                             int Bn, int C, int H, int W) {
  const int Ho = 2 * H, Wo = 2 * W;
  const unsigned total = (unsigned)Bn * Ho * Wo;
  const unsigned idx = blockIdx.x * blockDim.x + threadIdx.x;
  if (idx >= total) return;
  const int xo = (int)(idx % (unsigned)Wo); unsigned t = idx / (unsigned)Wo;
  const int yo = (int)(t % (unsigned)Ho);
  const int b  = (int)(t / (unsigned)Ho);
  const float sy = yo * 0.5f - 0.25f, sx = xo * 0.5f - 0.25f;
  int y0 = (int)floorf(sy), x0 = (int)floorf(sx);
  const float fy = sy - y0, fx = sx - x0;
  int y1 = y0 + 1, x1 = x0 + 1;
  y0 = min(max(y0, 0), H - 1); y1 = min(max(y1, 0), H - 1);
  x0 = min(max(x0, 0), W - 1); x1 = min(max(x1, 0), W - 1);
  const float w00 = (1 - fy) * (1 - fx), w01 = (1 - fy) * fx;
  const float w10 = fy * (1 - fx),       w11 = fy * fx;
  float acc = bias[0];
  for (int c = 0; c < C; ++c) {
    const float* p = x + ((long)b * C + c) * H * W;
    const float s = w00 * p[y0 * W + x0] + w01 * p[y0 * W + x1] +
                    w10 * p[y1 * W + x0] + w11 * p[y1 * W + x1];
    acc += s * w[c];
  }
  out[idx] = acc;
}

// ================================================================= host side
static inline int cdiv(long a, long b) { return (int)((a + b - 1) / b); }

static void gemm(hipStream_t s, const float* A, int lda, int amode, int L,
                 const float* W, int ldw, const float* bias, const float* resid,
                 float* out, int ldo, int omode, int M, int N, int K, int act) {
  const int tiles = cdiv(M, 16) * cdiv(N, 32);
  const int WPB = 4;  // waves per block
  const dim3 g(cdiv(tiles, WPB)), b(WPB * 32);
  if (amode == 0)
    k_gemm_wmma<0><<<g, b, 0, s>>>(A, lda, L, W, ldw, bias, resid, out, ldo,
                                   omode, M, N, K, act);
  else
    k_gemm_wmma<1><<<g, b, 0, s>>>(A, lda, L, W, ldw, bias, resid, out, ldo,
                                   omode, M, N, K, act);
}

// workspace layout (floats)
constexpr long F_X0  = 0;
constexpr long F_X1  = F_X0  + 4L * 64 * 64 * 64;
constexpr long F_X2  = F_X1  + 4L * 64 * 64 * 64;
constexpr long F_X3  = F_X2  + 4L * 128 * 32 * 32;
constexpr long F_X4  = F_X3  + 4L * 256 * 16 * 16;
constexpr long F_ID  = F_X4  + 4L * 512 * 8 * 8;
constexpr long F_H   = F_ID  + 4L * 128 * 32 * 32;
constexpr long F_XZ  = F_H   + 4L * 64 * 64 * 64;
constexpr long F_XH  = F_XZ  + 4L * 4096 * 256;
constexpr long F_DBC = F_XH  + 4L * 4096 * 128;
constexpr long F_DT  = F_DBC + 4L * 4096 * 132;
constexpr long F_Y   = F_DT  + 4L * 4096 * 128;
constexpr long F_UP  = F_Y   + 4L * 4096 * 128;
constexpr long F_MID = F_UP  + 4L * 64 * 64 * 64;
constexpr long F_D3  = F_MID + 4L * 64 * 64 * 64;
constexpr long F_D2  = F_D3  + 4L * 256 * 16 * 16;
constexpr long F_D1  = F_D2  + 4L * 128 * 32 * 32;

extern "C" void kernel_launch(void* const* d_in, const int* in_sizes, int n_in,
                              void* d_out, int out_size, void* d_ws, size_t ws_size,
                              hipStream_t stream) {
  (void)in_sizes; (void)n_in; (void)out_size; (void)ws_size;
#define IN(i) ((const float*)d_in[(i)])
  float* ws = (float*)d_ws;
  const int Bn = 4;
  const int T = 256;  // elementwise block size (8 waves)

  float* x0 = ws + F_X0; float* x1 = ws + F_X1; float* x2 = ws + F_X2;
  float* x3 = ws + F_X3; float* x4 = ws + F_X4;
  float* idb = ws + F_ID; float* hb = ws + F_H;
  float* xz = ws + F_XZ; float* xh = ws + F_XH; float* dbc = ws + F_DBC;
  float* dtb = ws + F_DT; float* yb = ws + F_Y;
  float* up = ws + F_UP; float* mid = ws + F_MID;
  float* d3 = ws + F_D3; float* d2 = ws + F_D2; float* d1 = ws + F_D1;

  // ---- stem: 7x7 s2 p3 conv + BN + ReLU : (4,3,128,128) -> (4,64,64,64)
  k_conv2d<<<cdiv(4L * 64 * 64 * 64, T), T, 0, stream>>>(
      IN(0), 3, nullptr, 0, IN(1), IN(2), IN(3), x0,
      Bn, 64, 128, 128, 64, 64, 7, 2, 3, 1);

  // ---- encoder blocks (conv+BN+ReLU -> Mamba -> +ident, ReLU)
  struct Enc { int base, proj, cin, d, Hin, stride; };
  const Enc encs[4] = { {4, -1, 64, 64, 64, 1}, {16, 28, 64, 128, 64, 2},
                        {31, 43, 128, 256, 32, 2}, {46, 58, 256, 512, 16, 2} };
  float* outs[4] = {x1, x2, x3, x4};
  const float* cur = x0;
  for (int i = 0; i < 4; ++i) {
    const Enc& E = encs[i];
    const int d = E.d, di = 2 * d, dr = (d + 15) / 16, ldd = dr + 128;
    const int Hout = E.Hin / E.stride, L = Hout * Hout, M = Bn * L;

    const float* ident = cur;
    if (E.proj >= 0) {  // 1x1 strided projection + BN (no relu)
      k_conv2d<<<cdiv((long)Bn * d * Hout * Hout, T), T, 0, stream>>>(
          cur, E.cin, nullptr, 0, IN(E.proj), IN(E.proj + 1), IN(E.proj + 2),
          idb, Bn, d, E.Hin, E.Hin, Hout, Hout, 1, E.stride, 0, 0);
      ident = idb;
    }
    // 3x3 conv + BN + ReLU -> h
    k_conv2d<<<cdiv((long)Bn * d * Hout * Hout, T), T, 0, stream>>>(
        cur, E.cin, nullptr, 0, IN(E.base), IN(E.base + 1), IN(E.base + 2),
        hb, Bn, d, E.Hin, E.Hin, Hout, Hout, 3, E.stride, 1, 1);

    // Mamba: xz = h * W_in^T   (h read NCHW-transposed by the GEMM)
    gemm(stream, hb, d, /*amode=*/1, L, IN(E.base + 3), d, nullptr, nullptr,
         xz, 2 * di, 0, M, 2 * di, d, 0);
    // causal depthwise conv1d + SiLU
    k_dwconv_silu<<<cdiv((long)M * di, T), T, 0, stream>>>(
        xz, IN(E.base + 4), IN(E.base + 5), xh, Bn, L, di);
    // dbc = xh * W_x^T
    gemm(stream, xh, di, 0, L, IN(E.base + 6), di, nullptr, nullptr,
         dbc, ldd, 0, M, ldd, di, 0);
    // dt = softplus(dbc[:, :dr] * W_dt^T + b_dt)   (fused bias+softplus)
    gemm(stream, dbc, ldd, 0, L, IN(E.base + 7), dr, IN(E.base + 8), nullptr,
         dtb, di, 0, M, di, dr, 3);
    // selective scan (state in VGPRs) + fused D-skip and SiLU(z) gate
    k_scan<<<cdiv((long)Bn * di * 32, T), T, 0, stream>>>(
        xh, dtb, dbc, xz, IN(E.base + 9), IN(E.base + 10), yb, Bn, L, di, dr);
    // out = relu( y * W_out^T  (transposed to NCHW)  + ident )  — all fused
    gemm(stream, yb, di, 0, L, IN(E.base + 11), di, nullptr, ident,
         outs[i], 0, /*omode=*/1, M, d, di, 1);
    cur = outs[i];
  }

  // ---- decoder blocks: upconv 2x2 s2, concat skip, two 3x3 conv+BN+ReLU
  struct Dec { int base, Ci, Co, H; const float* skip; float* out; };
  const Dec decs[3] = { {61, 512, 256, 8,  x3, d3},
                        {68, 256, 128, 16, x2, d2},
                        {75, 128, 64,  32, x1, d1} };
  for (int i = 0; i < 3; ++i) {
    const Dec& D = decs[i];
    const int H2 = 2 * D.H;
    k_upconv2x2<<<cdiv((long)Bn * D.Co * H2 * H2, T), T, 0, stream>>>(
        cur, IN(D.base), up, Bn, D.Ci, D.Co, D.H, D.H);
    k_conv2d<<<cdiv((long)Bn * D.Co * H2 * H2, T), T, 0, stream>>>(
        up, D.Co, D.skip, D.Co, IN(D.base + 1), IN(D.base + 2), IN(D.base + 3),
        mid, Bn, D.Co, H2, H2, H2, H2, 3, 1, 1, 1);
    k_conv2d<<<cdiv((long)Bn * D.Co * H2 * H2, T), T, 0, stream>>>(
        mid, D.Co, nullptr, 0, IN(D.base + 4), IN(D.base + 5), IN(D.base + 6),
        D.out, Bn, D.Co, H2, H2, H2, H2, 3, 1, 1, 1);
    cur = D.out;
  }

  // ---- fused bilinear 2x upsample + 1x1 seg conv + bias -> (4,1,128,128)
  k_resize_seg<<<cdiv(4L * 128 * 128, T), T, 0, stream>>>(
      d1, IN(82), IN(83), (float*)d_out, Bn, 64, 64, 64);
#undef IN
}
